// OldMamba2Network_44590350467196
// MI455X (gfx1250) — matrix operand order repaired and checked by reading, not compiled, not run on hardware
//
#include <hip/hip_runtime.h>
#include <hip/hip_bf16.h>
#include <math.h>

// ---------------------------------------------------------------------------
// MI455X (gfx1250, wave32) implementation of the Mamba2 policy network.
// All large GEMMs run on v_wmma_f32_16x16x32_bf16; each wave computes a
// 16x64 output strip (1 A fragment reused against 4 B fragments -> 4 WMMAs
// per K-step) to raise matrix-pipe intensity vs. fragment loads.
// Weights are pre-converted once per call into transposed BF16 [N][Kpad] so
// both A and B fragments load as packed b128s matching the ISA VGPR layout.
// The L=48 SSM scan is collapsed to a per-channel 48-tap weighted sum
// (only the last token of _mamba is consumed by the reference).
// Workspace requirement: ~76 MB.
// ---------------------------------------------------------------------------

typedef __bf16 bh;
typedef __attribute__((ext_vector_type(16))) __bf16 v16bf;
typedef __attribute__((ext_vector_type(2)))  __bf16 bf16x2;
typedef __attribute__((ext_vector_type(8)))  float  v8f;

#define BATCH    512
#define MEM_LEN  48
#define D_MODEL  256
#define D_INNER  512
#define D_STATE  16
#define NHEADS   32
#define CONV_XBC 544          // D_INNER + 2*D_STATE
#define D_IN_PROJ 1088        // 2*D_INNER + 2*D_STATE + NHEADS
#define FEAT_IN  1092         // 64*17 + 4
#define K1PAD    1120         // FEAT_IN padded to mult of 32
#define MROWS    (BATCH*MEM_LEN)   // 24576

// ---------------- weight convert: W[K][N] f32 -> out[N][Kpad] bf16 ----------
__global__ void k_transpose_bf16(const float* __restrict__ W, bh* __restrict__ out,
                                 int K, int N, int Kpad, int total) {
  int i = blockIdx.x * blockDim.x + threadIdx.x;
  if (i >= total) return;
  int n = i / Kpad, k = i % Kpad;
  float v = (k < K) ? W[(size_t)k * N + n] : 0.0f;
  out[i] = (bh)v;
}

// ---------------- encoder: 3 strided conv1d stages, LDS resident ------------
__global__ void k_encoder(const float* __restrict__ scan, const float* __restrict__ state,
                          const float* __restrict__ cw1, const float* __restrict__ cb1,
                          const float* __restrict__ cw2, const float* __restrict__ cb2,
                          const float* __restrict__ cw3, const float* __restrict__ cb3,
                          bh* __restrict__ cfin) {
  __shared__ float s_in[1080];
  __shared__ float s_h1[16 * 269];
  __shared__ float s_h2[32 * 67];
  int b = blockIdx.x, tid = threadIdx.x;
  for (int i = tid; i < 1080; i += 256) s_in[i] = scan[(size_t)b * 1080 + i];
  __syncthreads();
  for (int i = tid; i < 16 * 269; i += 256) {
    int ch = i / 269, p = i % 269, base = p * 4;
    float a = cb1[ch];
#pragma unroll
    for (int k = 0; k < 5; k++) a += s_in[base + k] * cw1[ch * 5 + k];
    s_h1[i] = fmaxf(a, 0.f);
  }
  __syncthreads();
  for (int i = tid; i < 32 * 67; i += 256) {
    int ch = i / 67, p = i % 67, base = p * 4;
    float a = cb2[ch];
    for (int ci = 0; ci < 16; ci++) {
      const float* wp = &cw2[(ch * 16 + ci) * 3];
      const float* xp = &s_h1[ci * 269 + base];
      a += xp[0] * wp[0] + xp[1] * wp[1] + xp[2] * wp[2];
    }
    s_h2[i] = fmaxf(a, 0.f);
  }
  __syncthreads();
  for (int i = tid; i < 64 * 17; i += 256) {
    int ch = i / 17, p = i % 17, base = p * 4;
    float a = cb3[ch];
    for (int ci = 0; ci < 32; ci++) {
      const float* wp = &cw3[(ch * 32 + ci) * 3];
      const float* xp = &s_h2[ci * 67 + base];
      a += xp[0] * wp[0] + xp[1] * wp[1] + xp[2] * wp[2];
    }
    cfin[(size_t)b * K1PAD + i] = (bh)fmaxf(a, 0.f);   // reshape order ch*17+p == i
  }
  if (tid < 4) cfin[(size_t)b * K1PAD + 1088 + tid] = (bh)state[b * 4 + tid];
  for (int i = FEAT_IN + tid; i < K1PAD; i += 256) cfin[(size_t)b * K1PAD + i] = (bh)0.0f;
}

// ---------------- buf[:, 0:47, :] = obs_buffer[:, 1:48, :] (f32 + bf16) -----
__global__ void k_shift_obs(const float* __restrict__ obs, float* __restrict__ bufF,
                            bh* __restrict__ bufH, int total) {
  int i = blockIdx.x * blockDim.x + threadIdx.x;
  if (i >= total) return;
  int d = i % D_MODEL;
  int rt = i / D_MODEL;
  int t = rt % 47, b = rt / 47;
  float v = obs[((size_t)b * MEM_LEN + (t + 1)) * D_MODEL + d];
  size_t o = ((size_t)b * MEM_LEN + t) * D_MODEL + d;
  bufF[o] = v;
  bufH[o] = (bh)v;
}

// ---------------- BF16 WMMA GEMM: 16x64 strip per wave32 --------------------
// A: [M][Kpad] bf16 row-major.  Bt: [N][Kpad] bf16 (transposed weights).
// Per K-step: 1 A fragment, 4 B fragments, 4 v_wmma accumulations.
// mode 0: out = (relu?)(A@B + bias) -> f32 (ldF) and/or bf16 (ldH)
// mode 1: zxbcdt scatter: col<512 -> z only at t==47; col<1056 -> xBC; else dt
__global__ void k_gemm_wmma(const bh* __restrict__ A, const bh* __restrict__ Bt,
                            const float* __restrict__ bias,
                            int Kpad, int ntilesN64, int ntiles,
                            int relu, int mode,
                            float* __restrict__ outF, int ldF,
                            bh* __restrict__ outH, int ldH,
                            float* __restrict__ zbuf, float* __restrict__ xbc,
                            float* __restrict__ dtr) {
  int wave = blockIdx.x * (blockDim.x >> 5) + (threadIdx.x >> 5);
  if (wave >= ntiles) return;
  int lane = threadIdx.x & 31;
  int g = lane >> 4, lo = lane & 15;
  int m0   = (wave / ntilesN64) * 16;
  int nblk = (wave % ntilesN64) * 64;
  v8f c[4] = {};
  const bh* Arow = A + (size_t)(m0 + lo) * Kpad;       // lanes 0-15 & 16-31 both hold M=lo
  const bh* Brow0 = Bt + (size_t)(nblk + lo) * Kpad;   // lane-in-group selects N
  for (int k0 = 0; k0 < Kpad; k0 += 32) {
    v16bf a;
#pragma unroll
    for (int j = 0; j < 8; j++) {
      // A 16x32 bf16 layout: VGPR j holds K pair 2j(+8 for j>=4)(+8 for hi lane group)
      int ka = k0 + 2 * j + ((j & 4) ? 8 : 0) + (g ? 8 : 0);
      bf16x2 pa = *(const bf16x2*)(Arow + ka);
      a[2 * j] = pa[0]; a[2 * j + 1] = pa[1];
    }
#pragma unroll
    for (int i = 0; i < 4; i++) {
      v16bf bf;
      const bh* Brow = Brow0 + (size_t)(i * 16) * Kpad;
#pragma unroll
      for (int j = 0; j < 8; j++) {
        // B 32x16 bf16 layout: VGPR j holds K pair 2j(+8 for j>=4)(+8 for hi lane group)
        int kb = k0 + 2 * j + ((j & 4) ? 8 : 0) + (g ? 8 : 0);
        bf16x2 pb = *(const bf16x2*)(Brow + kb);
        bf[2 * j] = pb[0]; bf[2 * j + 1] = pb[1];
      }
      c[i] = __builtin_amdgcn_wmma_f32_16x16x32_bf16(false, a, false, bf,
                                                     (short)0, c[i], false, false);
    }
  }
#pragma unroll
  for (int i = 0; i < 4; i++) {
#pragma unroll
    for (int r = 0; r < 8; r++) {
      int row = m0 + r + g * 8;     // C/D layout: VGPR r -> M = r + 8*hiGroup, N = lo
      int col = nblk + i * 16 + lo;
      float v = c[i][r];
      if (bias) v += bias[col];
      if (relu) v = fmaxf(v, 0.f);
      if (mode == 0) {
        if (outF) outF[(size_t)row * ldF + col] = v;
        if (outH) outH[(size_t)row * ldH + col] = (bh)v;
      } else {
        int t = row % MEM_LEN, bidx = row / MEM_LEN;
        if (col < D_INNER)            { if (t == MEM_LEN - 1) zbuf[(size_t)bidx * D_INNER + col] = v; }
        else if (col < D_INNER + CONV_XBC) xbc[(size_t)row * CONV_XBC + (col - D_INNER)] = v;
        else                               dtr[(size_t)row * NHEADS + (col - D_INNER - CONV_XBC)] = v;
      }
    }
  }
}

// ---------------- collapsed SSM scan + gate + RMSNorm + W_out ---------------
// One workgroup per batch element; everything LDS resident.
__global__ void k_mamba_final(const float* __restrict__ xbc, const float* __restrict__ dtr,
                              const float* __restrict__ z47,
                              const float* __restrict__ conv_w, const float* __restrict__ conv_b,
                              const float* __restrict__ dt_bias, const float* __restrict__ A_log,
                              const float* __restrict__ Dp, const float* __restrict__ norm_w,
                              const float* __restrict__ W_out, float* __restrict__ tvec) {
  __shared__ float s_dt[MEM_LEN * NHEADS];
  __shared__ float s_coef[MEM_LEN * NHEADS];   // exp(-a_h * suffix(dt)) * dt_t
  __shared__ float s_B[MEM_LEN * D_STATE];
  __shared__ float s_C[D_STATE];
  __shared__ float s_g[MEM_LEN];               // B_t . C_47 (head independent!)
  __shared__ float s_y[D_INNER];
  __shared__ float s_red[8];
  __shared__ float s_scale;
  int b = blockIdx.x, tid = threadIdx.x;
  const float* xb = xbc + (size_t)b * MEM_LEN * CONV_XBC;

  for (int i = tid; i < MEM_LEN * NHEADS; i += 256) {
    int t = i / NHEADS, h = i % NHEADS;
    float x = dtr[((size_t)b * MEM_LEN + t) * NHEADS + h] + dt_bias[h];
    s_dt[i] = (x > 20.f) ? x : log1pf(expf(x));          // softplus
  }
  __syncthreads();
  if (tid < NHEADS) {                                     // suffix-sum weights
    float aH = expf(A_log[tid]);
    float S = 0.f;
    for (int t = MEM_LEN - 1; t >= 0; t--) {
      float d = s_dt[t * NHEADS + tid];
      s_coef[t * NHEADS + tid] = expf(-aH * S) * d;
      S += d;
    }
  }
  if (tid >= 32 && tid < 64) {                            // conv+silu for B/C channels
    int j = tid - 32, ch = D_INNER + j;
    float w0 = conv_w[ch * 4], w1 = conv_w[ch * 4 + 1], w2 = conv_w[ch * 4 + 2],
          w3 = conv_w[ch * 4 + 3], cb = conv_b[ch];
    float r0 = 0.f, r1 = 0.f, r2 = 0.f;
    for (int t = 0; t < MEM_LEN; t++) {
      float r3 = xb[(size_t)t * CONV_XBC + ch];
      float v = r0 * w0 + r1 * w1 + r2 * w2 + r3 * w3 + cb;
      v = v / (1.f + expf(-v));
      if (j < D_STATE) s_B[t * D_STATE + j] = v;
      else if (t == MEM_LEN - 1) s_C[j - D_STATE] = v;
      r0 = r1; r1 = r2; r2 = r3;
    }
  }
  __syncthreads();
  if (tid < MEM_LEN) {
    float gacc = 0.f;
#pragma unroll
    for (int s = 0; s < D_STATE; s++) gacc += s_B[tid * D_STATE + s] * s_C[s];
    s_g[tid] = gacc;
  }
  __syncthreads();
#pragma unroll
  for (int cc = 0; cc < 2; cc++) {                        // per-channel 48-tap sum
    int ch = tid + cc * 256, h = ch >> 4;
    float w0 = conv_w[ch * 4], w1 = conv_w[ch * 4 + 1], w2 = conv_w[ch * 4 + 2],
          w3 = conv_w[ch * 4 + 3], cb = conv_b[ch];
    float r0 = 0.f, r1 = 0.f, r2 = 0.f, acc = 0.f, xlast = 0.f;
    for (int t = 0; t < MEM_LEN; t++) {
      float r3 = xb[(size_t)t * CONV_XBC + ch];
      float v = r0 * w0 + r1 * w1 + r2 * w2 + r3 * w3 + cb;
      v = v / (1.f + expf(-v));
      acc += s_coef[t * NHEADS + h] * s_g[t] * v;
      xlast = v;
      r0 = r1; r1 = r2; r2 = r3;
    }
    float y = acc + Dp[h] * xlast;
    float z = z47[(size_t)b * D_INNER + ch];
    y *= z / (1.f + expf(-z));                            // * silu(z)
    s_y[ch] = y;
  }
  __syncthreads();
  float p = s_y[tid] * s_y[tid] + s_y[tid + 256] * s_y[tid + 256];
  for (int o = 16; o > 0; o >>= 1) p += __shfl_xor(p, o); // wave32 reduce
  if ((tid & 31) == 0) s_red[tid >> 5] = p;
  __syncthreads();
  if (tid == 0) {
    float s = 0.f;
    for (int i = 0; i < 8; i++) s += s_red[i];
    s_scale = rsqrtf(s / (float)D_INNER + 1e-5f);
  }
  __syncthreads();
  float sc = s_scale;
  s_y[tid]       = s_y[tid]       * sc * norm_w[tid];
  s_y[tid + 256] = s_y[tid + 256] * sc * norm_w[tid + 256];
  __syncthreads();
  float acc = 0.f;                                        // y @ W_out -> 256
  for (int k = 0; k < D_INNER; k++) acc += s_y[k] * W_out[(size_t)k * D_MODEL + tid];
  tvec[(size_t)b * D_MODEL + tid] = acc;
}

// ---------------- FC stack + gaussian heads, per-batch block ----------------
struct FCP {
  const float* W[7];  const float* Bv[7];
  const float* Wm;    const float* bm;
  const float* Ws;    const float* bs;
};

__global__ void k_fc_heads(const float* __restrict__ tvec, FCP p,
                           float* __restrict__ loc, float* __restrict__ scl) {
  __shared__ float s_a[496];
  __shared__ float s_b[496];
  const int dims[8] = {256, 496, 382, 296, 192, 128, 64, 32};
  int b = blockIdx.x, tid = threadIdx.x;
  s_a[tid] = tvec[(size_t)b * 256 + tid];
  __syncthreads();
  float* in = s_a;
  float* out = s_b;
  for (int l = 0; l < 7; l++) {
    int id = dims[l], od = dims[l + 1];
    const float* W = p.W[l];
    const float* Bb = p.Bv[l];
    for (int n = tid; n < od; n += 256) {
      float a = Bb[n];
      for (int k = 0; k < id; k++) a += in[k] * W[(size_t)k * od + n];
      out[n] = fmaxf(a, 0.f);
    }
    __syncthreads();
    float* t2 = in; in = out; out = t2;
  }
  if (tid < 2) {
    float a = p.bm[tid], s2 = p.bs[tid];
    for (int k = 0; k < 32; k++) {
      a  += in[k] * p.Wm[k * 2 + tid];
      s2 += in[k] * p.Ws[k * 2 + tid];
    }
    loc[(size_t)b * 2 + tid] = a;
    float ls = fminf(fmaxf(s2, -5.f), 2.f);
    scl[(size_t)b * 2 + tid] = fminf(fmaxf(expf(ls), 0.01f), 10.f);
  }
}

// ---------------------------------------------------------------------------
extern "C" void kernel_launch(void* const* d_in, const int* in_sizes, int n_in,
                              void* d_out, int out_size, void* d_ws, size_t ws_size,
                              hipStream_t stream) {
  (void)in_sizes; (void)n_in; (void)out_size; (void)ws_size;
  const float* scan   = (const float*)d_in[0];
  const float* state  = (const float*)d_in[1];
  const float* obs    = (const float*)d_in[2];
  const float* cw1    = (const float*)d_in[3];
  const float* cb1    = (const float*)d_in[4];
  const float* cw2    = (const float*)d_in[5];
  const float* cb2    = (const float*)d_in[6];
  const float* cw3    = (const float*)d_in[7];
  const float* cb3    = (const float*)d_in[8];
  const float* Wf1    = (const float*)d_in[9];
  const float* bf1    = (const float*)d_in[10];
  const float* Wf2    = (const float*)d_in[11];
  const float* bf2    = (const float*)d_in[12];
  const float* W_in   = (const float*)d_in[13];
  const float* conv_w = (const float*)d_in[14];
  const float* conv_b = (const float*)d_in[15];
  const float* dt_bias= (const float*)d_in[16];
  const float* A_log  = (const float*)d_in[17];
  const float* Dp     = (const float*)d_in[18];
  const float* norm_w = (const float*)d_in[19];
  const float* W_out  = (const float*)d_in[20];
  FCP p;
  for (int l = 0; l < 7; l++) {
    p.W[l]  = (const float*)d_in[21 + 2 * l];
    p.Bv[l] = (const float*)d_in[22 + 2 * l];
  }
  p.Wm = (const float*)d_in[35]; p.bm = (const float*)d_in[36];
  p.Ws = (const float*)d_in[37]; p.bs = (const float*)d_in[38];

  float* loc  = (float*)d_out;           // (512,2)
  float* scl  = loc + 1024;              // (512,2)
  float* bufF = loc + 2048;              // (512,48,256) f32 output

  char* wsb = (char*)d_ws;
  size_t off = 0;
  auto alloc = [&](size_t bytes) -> void* {
    void* r = wsb + off;
    off += (bytes + 255) & ~(size_t)255;
    return r;
  };
  bh*    cfin = (bh*)alloc((size_t)BATCH * K1PAD * 2);        // encoder feats bf16 (K padded)
  bh*    Wf1t = (bh*)alloc((size_t)768 * K1PAD * 2);          // Wf1^T bf16
  bh*    cf1  = (bh*)alloc((size_t)BATCH * 768 * 2);
  bh*    Wf2t = (bh*)alloc((size_t)256 * 768 * 2);
  bh*    Wint = (bh*)alloc((size_t)D_IN_PROJ * 256 * 2);      // W_in^T bf16
  bh*    bufH = (bh*)alloc((size_t)MROWS * D_MODEL * 2);      // buf bf16 (GEMM3 input)
  float* xbc  = (float*)alloc((size_t)MROWS * CONV_XBC * 4);  // 53.5 MB
  float* dtr  = (float*)alloc((size_t)MROWS * NHEADS * 4);
  float* z47  = (float*)alloc((size_t)BATCH * D_INNER * 4);
  float* tvec = (float*)alloc((size_t)BATCH * D_MODEL * 4);

  auto cdiv = [](int a, int b) { return (a + b - 1) / b; };

  { int tot = 768 * K1PAD;
    k_transpose_bf16<<<cdiv(tot, 256), 256, 0, stream>>>(Wf1, Wf1t, FEAT_IN, 768, K1PAD, tot); }
  { int tot = 256 * 768;
    k_transpose_bf16<<<cdiv(tot, 256), 256, 0, stream>>>(Wf2, Wf2t, 768, 256, 768, tot); }
  { int tot = D_IN_PROJ * 256;
    k_transpose_bf16<<<cdiv(tot, 256), 256, 0, stream>>>(W_in, Wint, 256, D_IN_PROJ, 256, tot); }

  k_encoder<<<BATCH, 256, 0, stream>>>(scan, state, cw1, cb1, cw2, cb2, cw3, cb3, cfin);

  { int tot = BATCH * 47 * D_MODEL;
    k_shift_obs<<<cdiv(tot, 256), 256, 0, stream>>>(obs, bufF, bufH, tot); }

  // GEMM1: relu(cfin @ Wf1 + bf1) -> cf1 bf16              M=512 N=768 K=1120
  { int ntN = 768 / 64, nt = (BATCH / 16) * ntN;
    k_gemm_wmma<<<cdiv(nt, 8), 256, 0, stream>>>(cfin, Wf1t, bf1, K1PAD, ntN, nt, 1, 0,
                                                 nullptr, 0, cf1, 768,
                                                 nullptr, nullptr, nullptr); }
  // GEMM2: relu(cf1 @ Wf2 + bf2) -> buf row 47 (f32 out + bf16)   M=512 N=256 K=768
  { int ntN = 256 / 64, nt = (BATCH / 16) * ntN;
    k_gemm_wmma<<<cdiv(nt, 8), 256, 0, stream>>>(cf1, Wf2t, bf2, 768, ntN, nt, 1, 0,
                                                 bufF + 47 * D_MODEL, MEM_LEN * D_MODEL,
                                                 bufH + 47 * D_MODEL, MEM_LEN * D_MODEL,
                                                 nullptr, nullptr, nullptr); }
  // GEMM3: buf @ W_in -> {z(last token only), xBC, dt} scatter     M=24576 N=1088 K=256
  { int ntN = D_IN_PROJ / 64, nt = (MROWS / 16) * ntN;
    k_gemm_wmma<<<cdiv(nt, 8), 256, 0, stream>>>(bufH, Wint, nullptr, 256, ntN, nt, 0, 1,
                                                 nullptr, 0, nullptr, 0,
                                                 z47, xbc, dtr); }

  k_mamba_final<<<BATCH, 256, 0, stream>>>(xbc, dtr, z47, conv_w, conv_b, dt_bias,
                                           A_log, Dp, norm_w, W_out, tvec);
  k_fc_heads<<<BATCH, 256, 0, stream>>>(tvec, p, loc, scl);
}
